// PointCloudTransformer_88244398064373
// MI455X (gfx1250) — matrix operand order repaired and checked by reading, not compiled, not run on hardware
//
#include <hip/hip_runtime.h>

// ---------------------------------------------------------------------------
// PCT forward for MI455X (gfx1250, wave32).
// All GEMMs: v_wmma_f32_16x16x32_f16, fp16 operands pre-staged so every
// fragment is two b128 loads. Neighbor gather uses async global->LDS copies.
// ---------------------------------------------------------------------------

typedef _Float16 h16;
typedef __attribute__((ext_vector_type(16))) _Float16     v16h;
typedef __attribute__((ext_vector_type(8)))  _Float16     v8h;
typedef __attribute__((ext_vector_type(8)))  float        v8f;
typedef __attribute__((ext_vector_type(4)))  unsigned int u32x4;

static constexpr int   cB  = 16;
static constexpr int   cN  = 8192;
static constexpr int   cS1 = 512;
static constexpr int   cS2 = 256;
static constexpr int   cC1 = 64;
static constexpr int   cC2 = 128;
static constexpr int   cC3 = 256;
static constexpr int   cQK = 64;
static constexpr float BN_INV_C = 0.99999499987f;  // 1/sqrt(1+1e-5)

union Frag { v16h v; unsigned int u[8]; u32x4 q[2]; };
union H8   { u32x4 q; v8h h; };

__device__ __forceinline__ v8f wmma_f16(v16h a, v16h b, v8f c) {
    return __builtin_amdgcn_wmma_f32_16x16x32_f16(false, a, false, b, (short)0, c,
                                                  false, false);
}

// Async global -> LDS copy (16 bytes / lane), tracked by ASYNCcnt.
__device__ __forceinline__ void async_copy_b128(unsigned lds_addr,
                                                unsigned long long gaddr) {
    asm volatile("global_load_async_to_lds_b128 %0, %1, off"
                 :: "v"(lds_addr), "v"(gaddr) : "memory");
}
__device__ __forceinline__ void wait_async0() {
    asm volatile("s_wait_asynccnt 0" ::: "memory");
}

// A fragment: 16x32 tile, row-major h16, ld in elements (mult of 8).
// Lane layout (ISA 7.12.2): m=lane&15, hi=lane>>4; VGPRs 0..3 hold halves
// [hi*8, hi*8+8), VGPRs 4..7 hold halves [16+hi*8, 16+hi*8+8) -> 2x b128.
__device__ __forceinline__ v16h load_a_frag(const h16* base, int ld, int lane) {
    int m = lane & 15, hi = lane >> 4;
    const h16* p = base + m * ld + hi * 8;
    Frag f;
    f.q[0] = *(const u32x4*)(p);
    f.q[1] = *(const u32x4*)(p + 16);
    return f.v;
}

// B fragment: 32x16 tile where the source is row-major [N][K] h16 (transposed
// weights, K-matrix, V^T). Lane n=lane&15, hi=lane>>4; VGPRs 0..3 hold halves
// [hi*8, hi*8+8), VGPRs 4..7 hold halves [16+hi*8, 16+hi*8+8) -> 2x b128.
__device__ __forceinline__ v16h load_b_frag_t(const h16* Wt, int ld, int lane) {
    int n = lane & 15, hi = lane >> 4;
    const h16* p = Wt + n * ld + hi * 8;
    Frag f;
    f.q[0] = *(const u32x4*)(p);
    f.q[1] = *(const u32x4*)(p + 16);
    return f.v;
}

// A fragment of (x - xr), both row-major h16: packed half-vector subtract.
__device__ __forceinline__ v16h load_a_frag_diff(const h16* x, const h16* xr,
                                                 int ld, int lane) {
    int m = lane & 15, hi = lane >> 4;
    const h16* px = x + m * ld + hi * 8;
    const h16* pr = xr + m * ld + hi * 8;
    H8 a0, a1, r0, r1, d0, d1;
    a0.q = *(const u32x4*)(px);      a1.q = *(const u32x4*)(px + 16);
    r0.q = *(const u32x4*)(pr);      r1.q = *(const u32x4*)(pr + 16);
    d0.h = a0.h - r0.h;              d1.h = a1.h - r1.h;
    Frag f;
    f.q[0] = d0.q;                   f.q[1] = d1.q;
    return f.v;
}

// ---------------------------------------------------------------------------
// Kernel 0: weight prep — fp32 [K][N] -> fp16 transposed [N][K].
// ---------------------------------------------------------------------------
__global__ void k_wprep(const float* __restrict__ W, int K, int N,
                        h16* __restrict__ Wt) {
    int i = blockIdx.x * blockDim.x + threadIdx.x;
    if (i < K * N) {
        int k = i / N, n = i % N;
        Wt[(size_t)n * K + k] = (h16)W[i];
    }
}

// ---------------------------------------------------------------------------
// Kernel 1: fused LBR1 (3->64, VALU) + LBR2 (64->64, WMMA) -> feat1 fp16
// grid = B*N/128, block = 256 (8 waves x 16 points)
// ---------------------------------------------------------------------------
__global__ void k_lbr12(const float* __restrict__ xyz,
                        const float* W1, const float* b1, const float* be1, const float* ga1,
                        const h16* __restrict__ W2t,  // [64][64] transposed
                        const float* b2, const float* be2, const float* ga2,
                        h16* __restrict__ feat_out) {
    __shared__ __align__(16) h16 h1[8][16 * 64];
    int wave = threadIdx.x >> 5, lane = threadIdx.x & 31;
    int base = blockIdx.x * 128 + wave * 16;
    int p = lane & 15, cbase = (lane >> 4) * 32;
    float x0 = xyz[(size_t)(base + p) * 3 + 0];
    float x1 = xyz[(size_t)(base + p) * 3 + 1];
    float x2 = xyz[(size_t)(base + p) * 3 + 2];
#pragma unroll 4
    for (int c = 0; c < 32; ++c) {
        int ch = cbase + c;
        float y = x0 * W1[0 * 64 + ch] + x1 * W1[1 * 64 + ch] + x2 * W1[2 * 64 + ch] + b1[ch];
        y = fmaxf(y * (ga1[ch] * BN_INV_C) + be1[ch], 0.f);
        h1[wave][p * 64 + ch] = (h16)y;
    }
    __syncthreads();
    v16h a0 = load_a_frag(&h1[wave][0], 64, lane);
    v16h a1 = load_a_frag(&h1[wave][32], 64, lane);
#pragma unroll
    for (int nt = 0; nt < 4; ++nt) {
        v8f acc = {};
        acc = wmma_f16(a0, load_b_frag_t(W2t + nt * 16 * 64, 64, lane), acc);
        acc = wmma_f16(a1, load_b_frag_t(W2t + nt * 16 * 64 + 32, 64, lane), acc);
        int n = nt * 16 + (lane & 15), hi = lane >> 4;
        float gm = ga2[n] * BN_INV_C, bt = be2[n], bi = b2[n];
#pragma unroll
        for (int r = 0; r < 8; ++r) {
            int m = r + 8 * hi;
            float y = fmaxf((acc[r] + bi) * gm + bt, 0.f);
            feat_out[(size_t)(base + m) * 64 + n] = (h16)y;
        }
    }
}

// ---------------------------------------------------------------------------
// Kernel 2: farthest point sampling. grid = B, block = 1024.
// ---------------------------------------------------------------------------
__global__ void __launch_bounds__(1024)
k_fps(const h16* __restrict__ feat, int N, int C, int S,
      int* __restrict__ idx_out, float* __restrict__ mind) {
    int b = blockIdx.x;
    const h16* f = feat + (size_t)b * N * C;
    float* md = mind + (size_t)b * N;
    __shared__ float cfeat[256];
    __shared__ float rval[1024];
    __shared__ int   ridx[1024];
    __shared__ int   cur;
    int t = threadIdx.x, nt = blockDim.x;
    if (t == 0) { cur = 0; idx_out[(size_t)b * S] = 0; }
    __syncthreads();
    for (int c = t; c < C; c += nt) cfeat[c] = (float)f[c];
    __syncthreads();
    for (int i = t; i < N; i += nt) {
        float d = 0.f;
        for (int c = 0; c < C; ++c) { float v = (float)f[(size_t)i * C + c] - cfeat[c]; d += v * v; }
        md[i] = d;
    }
    for (int s = 1; s < S; ++s) {
        __syncthreads();
        float bv = -1.f; int bi = 0;
        for (int i = t; i < N; i += nt) { float v = md[i]; if (v > bv) { bv = v; bi = i; } }
        rval[t] = bv; ridx[t] = bi;
        __syncthreads();
        for (int off = nt >> 1; off > 0; off >>= 1) {
            if (t < off && rval[t + off] > rval[t]) { rval[t] = rval[t + off]; ridx[t] = ridx[t + off]; }
            __syncthreads();
        }
        if (t == 0) { cur = ridx[0]; idx_out[(size_t)b * S + s] = cur; }
        __syncthreads();
        for (int c = t; c < C; c += nt) cfeat[c] = (float)f[(size_t)cur * C + c];
        __syncthreads();
        for (int i = t; i < N; i += nt) {
            float d = 0.f;
            for (int c = 0; c < C; ++c) { float v = (float)f[(size_t)i * C + c] - cfeat[c]; d += v * v; }
            float m = md[i]; md[i] = d < m ? d : m;
        }
    }
}

// ---------------------------------------------------------------------------
// Kernel 3: kNN (k=32). grid = B*S, block = 256.
// ---------------------------------------------------------------------------
__global__ void k_knn(const h16* __restrict__ feat, int N, int C, int S,
                      const int* __restrict__ fps_idx, int* __restrict__ knn_out) {
    __shared__ float dist[8192];
    __shared__ float cfeat[256];
    __shared__ float rval[256];
    __shared__ int   ridx[256];
    int bs = blockIdx.x, b = bs / S, s = bs % S;
    const h16* f = feat + (size_t)b * N * C;
    int t = threadIdx.x, nt = blockDim.x;
    int ci = fps_idx[(size_t)b * S + s];
    for (int c = t; c < C; c += nt) cfeat[c] = (float)f[(size_t)ci * C + c];
    __syncthreads();
    for (int i = t; i < N; i += nt) {
        float d = 0.f;
        for (int c = 0; c < C; ++c) { float v = (float)f[(size_t)i * C + c] - cfeat[c]; d += v * v; }
        dist[i] = d;
    }
    __syncthreads();
    for (int k = 0; k < 32; ++k) {
        float bv = 3.4e38f; int bi = 0;
        for (int i = t; i < N; i += nt) { float v = dist[i]; if (v < bv) { bv = v; bi = i; } }
        rval[t] = bv; ridx[t] = bi;
        __syncthreads();
        for (int off = nt >> 1; off > 0; off >>= 1) {
            if (t < off && rval[t + off] < rval[t]) { rval[t] = rval[t + off]; ridx[t] = ridx[t + off]; }
            __syncthreads();
        }
        if (t == 0) { knn_out[((size_t)b * S + s) * 32 + k] = ridx[0]; dist[ridx[0]] = 3.4e38f; }
        __syncthreads();
    }
}

// ---------------------------------------------------------------------------
// Kernel 4: sample-and-group LBR1+LBR2+maxpool. grid = B*S, block = 256.
// Neighbor rows staged via async global->LDS, then two WMMA layers, max over k.
// LDS pool phase-reused: [g | nstage->h1s | h2s], 48 KB.
// ---------------------------------------------------------------------------
__global__ void k_sg(const h16* __restrict__ feat_in, int Nin, int C, int Cout, int S,
                     const int* __restrict__ fps_idx, const int* __restrict__ knn_idx,
                     const h16* __restrict__ W1t,  // [Cout][Cin] transposed fp16
                     const float* b1, const float* be1, const float* ga1,
                     const h16* __restrict__ W2t,  // [Cout][Cout] transposed fp16
                     const float* b2, const float* be2, const float* ga2,
                     h16* __restrict__ feat_out) {
    __shared__ __align__(16) h16 pool[3 * 32 * 256];
    __shared__ float ctr[128];
    h16* g      = pool;                 // [32][Cin]
    h16* h1s    = pool + 32 * 256;      // layer-1 activations (aliases nstage)
    h16* h2s    = pool + 2 * 32 * 256;  // layer-2 activations
    h16* nstage = h1s;                  // raw neighbor rows [32][C]

    int bs = blockIdx.x, b = bs / S, s = bs % S;
    const h16* f = feat_in + (size_t)b * Nin * C;
    int t = threadIdx.x, wave = t >> 5, lane = t & 31;
    int Cin = 2 * C;
    int ci = fps_idx[(size_t)b * S + s];
    for (int c = t; c < C; c += 256) ctr[c] = (float)f[(size_t)ci * C + c];

    // Async-stage the 32 neighbor rows (16B chunks per lane, ASYNCcnt tracked).
    const int* nbr = knn_idx + ((size_t)b * S + s) * 32;
    int cpr = C >> 3;                   // 16-byte chunks per row
    for (int e = t; e < 32 * cpr; e += 256) {
        int kk = e / cpr, ch = e % cpr;
        int ni = nbr[kk];
        unsigned long long ga = (unsigned long long)(f + (size_t)ni * C + ch * 8);
        unsigned la = (unsigned)(unsigned long long)(nstage + kk * C + ch * 8);
        async_copy_b128(la, ga);
    }
    wait_async0();
    __syncthreads();

    // Build g = [nbr - ctr, ctr]
    for (int e = t; e < 32 * C; e += 256) {
        int kk = e / C, c = e % C;
        float v = (float)nstage[kk * C + c];
        g[kk * Cin + c]     = (h16)(v - ctr[c]);
        g[kk * Cin + C + c] = (h16)ctr[c];
    }
    __syncthreads();

    int NT = Cout >> 4;
    for (int ntile = wave; ntile < NT; ntile += 8) {
        for (int mr = 0; mr < 2; ++mr) {
            v8f acc = {};
            for (int kc = 0; kc < Cin; kc += 32) {
                v16h a  = load_a_frag(g + mr * 16 * Cin + kc, Cin, lane);
                v16h bf = load_b_frag_t(W1t + (size_t)ntile * 16 * Cin + kc, Cin, lane);
                acc = wmma_f16(a, bf, acc);
            }
            int n = ntile * 16 + (lane & 15), hi = lane >> 4;
            float gm = ga1[n] * BN_INV_C, bt = be1[n], bi = b1[n];
#pragma unroll
            for (int r = 0; r < 8; ++r) {
                int m = mr * 16 + r + 8 * hi;
                h1s[m * Cout + n] = (h16)fmaxf((acc[r] + bi) * gm + bt, 0.f);
            }
        }
    }
    __syncthreads();
    for (int ntile = wave; ntile < NT; ntile += 8) {
        for (int mr = 0; mr < 2; ++mr) {
            v8f acc = {};
            for (int kc = 0; kc < Cout; kc += 32) {
                v16h a  = load_a_frag(h1s + mr * 16 * Cout + kc, Cout, lane);
                v16h bf = load_b_frag_t(W2t + (size_t)ntile * 16 * Cout + kc, Cout, lane);
                acc = wmma_f16(a, bf, acc);
            }
            int n = ntile * 16 + (lane & 15), hi = lane >> 4;
            float gm = ga2[n] * BN_INV_C, bt = be2[n], bi = b2[n];
#pragma unroll
            for (int r = 0; r < 8; ++r) {
                int m = mr * 16 + r + 8 * hi;
                h2s[m * Cout + n] = (h16)fmaxf((acc[r] + bi) * gm + bt, 0.f);
            }
        }
    }
    __syncthreads();
    for (int c = t; c < Cout; c += 256) {
        float m = 0.f;
        for (int kk = 0; kk < 32; ++kk) { float v = (float)h2s[kk * Cout + c]; m = v > m ? v : m; }
        feat_out[((size_t)b * S + s) * Cout + c] = (h16)m;
    }
}

// ---------------------------------------------------------------------------
// Kernel 5: projection out = A @ Wt^T + bias (fp16 out, row-major [M][N]).
// block = 128 (4 waves), grid = (M/16, N/64).
// ---------------------------------------------------------------------------
__global__ void k_gemm_proj(const h16* __restrict__ A, const h16* __restrict__ Wt,
                            const float* __restrict__ bias,
                            int M, int Kd, int N, h16* __restrict__ out) {
    int wave = threadIdx.x >> 5, lane = threadIdx.x & 31;
    int m0 = blockIdx.x * 16;
    int ntile = blockIdx.y * 4 + wave;
    v8f acc = {};
    for (int kc = 0; kc < Kd; kc += 32) {
        __builtin_prefetch(A + (size_t)m0 * Kd + kc + 64, 0, 1);
        v16h a  = load_a_frag(A + (size_t)m0 * Kd + kc, Kd, lane);
        v16h bf = load_b_frag_t(Wt + (size_t)ntile * 16 * Kd + kc, Kd, lane);
        acc = wmma_f16(a, bf, acc);
    }
    int n = ntile * 16 + (lane & 15), hi = lane >> 4;
    float bi = bias[n];
#pragma unroll
    for (int r = 0; r < 8; ++r) {
        int m = m0 + r + 8 * hi;
        out[(size_t)m * N + n] = (h16)(acc[r] + bi);
    }
}

// Variant writing the result transposed per batch: out_t[b][n][s] (for V^T).
__global__ void k_gemm_proj_t(const h16* __restrict__ A, const h16* __restrict__ Wt,
                              const float* __restrict__ bias,
                              int M, int Kd, int N, int Sdim,
                              h16* __restrict__ out_t) {
    int wave = threadIdx.x >> 5, lane = threadIdx.x & 31;
    int m0 = blockIdx.x * 16;
    int ntile = blockIdx.y * 4 + wave;
    v8f acc = {};
    for (int kc = 0; kc < Kd; kc += 32) {
        __builtin_prefetch(A + (size_t)m0 * Kd + kc + 64, 0, 1);
        v16h a  = load_a_frag(A + (size_t)m0 * Kd + kc, Kd, lane);
        v16h bf = load_b_frag_t(Wt + (size_t)ntile * 16 * Kd + kc, Kd, lane);
        acc = wmma_f16(a, bf, acc);
    }
    int n = ntile * 16 + (lane & 15), hi = lane >> 4;
    float bi = bias[n];
#pragma unroll
    for (int r = 0; r < 8; ++r) {
        int m = m0 + r + 8 * hi;
        int bb = m / Sdim, ss = m % Sdim;
        out_t[((size_t)bb * N + n) * Sdim + ss] = (h16)(acc[r] + bi);
    }
}

// ---------------------------------------------------------------------------
// Kernel 6: attention scores S = Q @ K^T. grid = (B, S/16, S/16), block = 32.
// K is [N][K]-shaped for the B fragment -> b128 loads.
// ---------------------------------------------------------------------------
__global__ void k_scores(const h16* __restrict__ q, const h16* __restrict__ k,
                         int S, int D, float* __restrict__ sc) {
    int b = blockIdx.x, mt = blockIdx.y, ntile = blockIdx.z, lane = threadIdx.x & 31;
    const h16* qb = q + (size_t)b * S * D;
    const h16* kb = k + (size_t)b * S * D;
    v8f acc = {};
    for (int kc = 0; kc < D; kc += 32) {
        v16h a  = load_a_frag(qb + (size_t)mt * 16 * D + kc, D, lane);
        v16h bf = load_b_frag_t(kb + (size_t)ntile * 16 * D + kc, D, lane);
        acc = wmma_f16(a, bf, acc);
    }
    int n = ntile * 16 + (lane & 15), hi = lane >> 4;
#pragma unroll
    for (int r = 0; r < 8; ++r) {
        int m = mt * 16 + r + 8 * hi;
        sc[((size_t)b * S + m) * S + n] = acc[r];
    }
}

// ---------------------------------------------------------------------------
// Kernel 7: row softmax in place. grid = B*S, block = 256 (= S).
// ---------------------------------------------------------------------------
__global__ void k_softmax_row(float* __restrict__ sc, int S) {
    int row = blockIdx.x, t = threadIdx.x;
    float* r = sc + (size_t)row * S;
    __shared__ float red[256];
    float v = (t < S) ? r[t] : -3.4e38f;
    red[t] = v; __syncthreads();
    for (int off = 128; off > 0; off >>= 1) {
        if (t < off) red[t] = fmaxf(red[t], red[t + off]);
        __syncthreads();
    }
    float mx = red[0]; __syncthreads();
    float e = (t < S) ? __expf(v - mx) : 0.f;
    red[t] = e; __syncthreads();
    for (int off = 128; off > 0; off >>= 1) {
        if (t < off) red[t] += red[t + off];
        __syncthreads();
    }
    if (t < S) r[t] = e / red[0];
}

// ---------------------------------------------------------------------------
// Kernel 8: column L1 renorm + fp16 conversion. grid = B, block = 256.
// ---------------------------------------------------------------------------
__global__ void k_l1(const float* __restrict__ sc, h16* __restrict__ a_out, int S) {
    int b = blockIdx.x, c = threadIdx.x;
    const float* base = sc + (size_t)b * S * S;
    float sum = 0.f;
    for (int n = 0; n < S; ++n) sum += base[(size_t)n * S + c];
    float inv = 1.f / (1e-9f + sum);
    for (int n = 0; n < S; ++n)
        a_out[((size_t)b * S + n) * S + c] = (h16)(base[(size_t)n * S + c] * inv);
}

// ---------------------------------------------------------------------------
// Kernel 9: x_r = A @ V using V^T [B][C][S]. grid = (B, S/16, C/16), block = 32.
// ---------------------------------------------------------------------------
__global__ void k_av(const h16* __restrict__ a, const h16* __restrict__ vt,
                     int S, int C, h16* __restrict__ xr) {
    int b = blockIdx.x, mt = blockIdx.y, ntile = blockIdx.z, lane = threadIdx.x & 31;
    const h16* ab  = a  + (size_t)b * S * S;
    const h16* vtb = vt + ((size_t)b * C + ntile * 16) * S;
    v8f acc = {};
    for (int kc = 0; kc < S; kc += 32) {
        v16h af = load_a_frag(ab + (size_t)mt * 16 * S + kc, S, lane);
        v16h bf = load_b_frag_t(vtb + kc, S, lane);
        acc = wmma_f16(af, bf, acc);
    }
    int n = ntile * 16 + (lane & 15), hi = lane >> 4;
#pragma unroll
    for (int r = 0; r < 8; ++r) {
        int m = mt * 16 + r + 8 * hi;
        xr[((size_t)b * S + m) * C + n] = (h16)acc[r];
    }
}

// ---------------------------------------------------------------------------
// Kernel 10: out = x + LBR(x - x_r). grid = (M/16, C/64), block = 128.
// ---------------------------------------------------------------------------
__global__ void k_oa_final(const h16* __restrict__ x, const h16* __restrict__ xr,
                           const h16* __restrict__ Wt, const float* bias,
                           const float* beta, const float* gamma,
                           int M, int C, h16* __restrict__ x_next,
                           float* __restrict__ out, int out_ld, int out_off) {
    int wave = threadIdx.x >> 5, lane = threadIdx.x & 31;
    int m0 = blockIdx.x * 16;
    int ntile = blockIdx.y * 4 + wave;
    v8f acc = {};
    for (int kc = 0; kc < C; kc += 32) {
        v16h a  = load_a_frag_diff(x + (size_t)m0 * C + kc, xr + (size_t)m0 * C + kc, C, lane);
        v16h bf = load_b_frag_t(Wt + (size_t)ntile * 16 * C + kc, C, lane);
        acc = wmma_f16(a, bf, acc);
    }
    int n = ntile * 16 + (lane & 15), hi = lane >> 4;
    float gm = gamma[n] * BN_INV_C, bt = beta[n], bi = bias[n];
#pragma unroll
    for (int r = 0; r < 8; ++r) {
        int m = m0 + r + 8 * hi;
        float t = fmaxf((acc[r] + bi) * gm + bt, 0.f);
        float res = (float)x[(size_t)m * C + n] + t;
        x_next[(size_t)m * C + n] = (h16)res;
        out[(size_t)m * out_ld + out_off + n] = res;
    }
}

// ---------------------------------------------------------------------------
// Host-side orchestration
// ---------------------------------------------------------------------------
extern "C" void kernel_launch(void* const* d_in, const int* in_sizes, int n_in,
                              void* d_out, int out_size, void* d_ws, size_t ws_size,
                              hipStream_t stream) {
    (void)in_sizes; (void)n_in; (void)out_size; (void)ws_size;
    const float* xyz = (const float*)d_in[0];
    auto P = [&](int i) { return (const float*)d_in[i]; };
    // leaf order (jax pytree, dicts sorted by key):
    // 0 inputs; lbr1: 1 W,2 b,3 beta,4 gamma; lbr2: 5..8;
    // oa[j] base 9+10j: Wk,Wq,Wv,bk,bq,bv,lbr.W,lbr.b,lbr.beta,lbr.gamma;
    // sg1.lbr1: 49..52; sg1.lbr2: 53..56; sg2.lbr1: 57..60; sg2.lbr2: 61..64; 65 is_training

    char* ws = (char*)d_ws;
    size_t off = 0;
    auto alloc = [&](size_t bytes) -> void* {
        void* p = ws + off; off += (bytes + 255) & ~(size_t)255; return p;
    };
    h16*   feat1 = (h16*)  alloc((size_t)cB * cN  * cC1 * 2);
    float* mind  = (float*)alloc((size_t)cB * cN  * 4);
    int*   fps1  = (int*)  alloc((size_t)cB * cS1 * 4);
    int*   knn1  = (int*)  alloc((size_t)cB * cS1 * 32 * 4);
    h16*   feat2 = (h16*)  alloc((size_t)cB * cS1 * cC2 * 2);
    int*   fps2  = (int*)  alloc((size_t)cB * cS2 * 4);
    int*   knn2  = (int*)  alloc((size_t)cB * cS2 * 32 * 4);
    h16*   x0    = (h16*)  alloc((size_t)cB * cS2 * cC3 * 2);
    h16*   x1    = (h16*)  alloc((size_t)cB * cS2 * cC3 * 2);
    h16*   qb    = (h16*)  alloc((size_t)cB * cS2 * cQK * 2);
    h16*   kb    = (h16*)  alloc((size_t)cB * cS2 * cQK * 2);
    h16*   vt    = (h16*)  alloc((size_t)cB * cC3 * cS2 * 2);
    h16*   xr    = (h16*)  alloc((size_t)cB * cS2 * cC3 * 2);
    float* sc    = (float*)alloc((size_t)cB * cS2 * cS2 * 4);
    h16*   ab    = (h16*)  alloc((size_t)cB * cS2 * cS2 * 2);
    // transposed fp16 weights
    h16* w2t    = (h16*)alloc((size_t)64 * 64 * 2);
    h16* sg1w1t = (h16*)alloc((size_t)128 * 128 * 2);
    h16* sg1w2t = (h16*)alloc((size_t)128 * 128 * 2);
    h16* sg2w1t = (h16*)alloc((size_t)256 * 256 * 2);
    h16* sg2w2t = (h16*)alloc((size_t)256 * 256 * 2);
    h16 *wqt[4], *wkt[4], *wvt[4], *lwt[4];
    for (int j = 0; j < 4; ++j) {
        wqt[j] = (h16*)alloc((size_t)64 * 256 * 2);
        wkt[j] = (h16*)alloc((size_t)64 * 256 * 2);
        wvt[j] = (h16*)alloc((size_t)256 * 256 * 2);
        lwt[j] = (h16*)alloc((size_t)256 * 256 * 2);
    }

    auto prep = [&](const float* W, int K, int N, h16* Wt) {
        int tot = K * N;
        k_wprep<<<(tot + 255) / 256, 256, 0, stream>>>(W, K, N, Wt);
    };
    prep(P(5), 64, 64, w2t);
    prep(P(49), 128, 128, sg1w1t);
    prep(P(53), 128, 128, sg1w2t);
    prep(P(57), 256, 256, sg2w1t);
    prep(P(61), 256, 256, sg2w2t);
    for (int j = 0; j < 4; ++j) {
        int base = 9 + j * 10;
        prep(P(base + 1), 256, 64, wqt[j]);   // Wq
        prep(P(base + 0), 256, 64, wkt[j]);   // Wk
        prep(P(base + 2), 256, 256, wvt[j]);  // Wv
        prep(P(base + 6), 256, 256, lwt[j]);  // lbr.W
    }

    // Stage 1: pointwise LBRs
    k_lbr12<<<cB * cN / 128, 256, 0, stream>>>(xyz, P(1), P(2), P(3), P(4),
                                               w2t, P(6), P(7), P(8), feat1);
    // Stage 2: SG1
    k_fps<<<cB, 1024, 0, stream>>>(feat1, cN, cC1, cS1, fps1, mind);
    k_knn<<<cB * cS1, 256, 0, stream>>>(feat1, cN, cC1, cS1, fps1, knn1);
    k_sg<<<cB * cS1, 256, 0, stream>>>(feat1, cN, cC1, cC2, cS1, fps1, knn1,
                                       sg1w1t, P(50), P(51), P(52),
                                       sg1w2t, P(54), P(55), P(56), feat2);
    // Stage 3: SG2
    k_fps<<<cB, 1024, 0, stream>>>(feat2, cS1, cC2, cS2, fps2, mind);
    k_knn<<<cB * cS2, 256, 0, stream>>>(feat2, cS1, cC2, cS2, fps2, knn2);
    k_sg<<<cB * cS2, 256, 0, stream>>>(feat2, cS1, cC2, cC3, cS2, fps2, knn2,
                                       sg2w1t, P(58), P(59), P(60),
                                       sg2w2t, P(62), P(63), P(64), x0);
    // Stage 4: 4 offset-attention blocks
    float* out = (float*)d_out;
    const int M = cB * cS2;  // 4096
    h16* xin = x0; h16* xout = x1;
    for (int blk = 0; blk < 4; ++blk) {
        int base = 9 + blk * 10;
        const float *bk = P(base + 3), *bq = P(base + 4), *bv = P(base + 5);
        const float *lb = P(base + 7), *lbe = P(base + 8), *lga = P(base + 9);
        k_gemm_proj<<<dim3(M / 16, 1), 128, 0, stream>>>(xin, wqt[blk], bq, M, cC3, cQK, qb);
        k_gemm_proj<<<dim3(M / 16, 1), 128, 0, stream>>>(xin, wkt[blk], bk, M, cC3, cQK, kb);
        k_gemm_proj_t<<<dim3(M / 16, 4), 128, 0, stream>>>(xin, wvt[blk], bv, M, cC3, cC3, cS2, vt);
        k_scores<<<dim3(cB, cS2 / 16, cS2 / 16), 32, 0, stream>>>(qb, kb, cS2, cQK, sc);
        k_softmax_row<<<cB * cS2, 256, 0, stream>>>(sc, cS2);
        k_l1<<<cB, 256, 0, stream>>>(sc, ab, cS2);
        k_av<<<dim3(cB, cS2 / 16, cC3 / 16), 32, 0, stream>>>(ab, vt, cS2, cC3, xr);
        k_oa_final<<<dim3(M / 16, 4), 128, 0, stream>>>(xin, xr, lwt[blk], lb, lbe, lga,
                                                        M, cC3, xout, out,
                                                        4 * cC3, blk * cC3);
        h16* tmp = xin; xin = xout; xout = tmp;
    }
}
